// MLP_TI_Gram_31258771980321
// MI455X (gfx1250) — matrix operand order repaired and checked by reading, not compiled, not run on hardware
//
#include <hip/hip_runtime.h>
#include <hip/hip_bf16.h>

typedef __attribute__((ext_vector_type(16))) _Float16 v16h;
typedef __attribute__((ext_vector_type(8)))  float    v8f;

#define N_PTS 1024
#define BATCH 128
#define HID   128
#define SREP  2176   // 64 front pad + 1024 signal + 1088 wrap tail

union Frag {
    v16h h;
    uint4 q[2];
    unsigned u[8];
};

__device__ __forceinline__ float celu1(float v) {
    // jax.nn.celu with alpha=1: x>0 ? x : exp(x)-1
    return v > 0.0f ? v : expm1f(v);
}

// Low 32 bits of a generic pointer into __shared__ == wave-relative LDS byte
// offset (ISA: LDS aperture uses addr[31:0]); that is what the async-to-LDS
// instruction's VDST operand expects.
__device__ __forceinline__ unsigned lds_off32(const void* p) {
    return (unsigned)(uintptr_t)p;
}

// -----------------------------------------------------------------------------
// Kernel 0: weight prep. W1T[n][k] = (f16)W1[k][n], W2T[n][k] = (f16)W2[k][n].
// Puts the GEMM B-operands in k-major f16 rows so WMMA B-fragments become two
// contiguous global_load_b128 per tile.
// -----------------------------------------------------------------------------
__global__ __launch_bounds__(256)
void prep_kernel(const float* __restrict__ W1, const float* __restrict__ W2,
                 _Float16* __restrict__ W1T, _Float16* __restrict__ W2T) {
    const int stride = gridDim.x * blockDim.x;
    int idx = blockIdx.x * blockDim.x + threadIdx.x;
    for (int p = idx; p < HID * N_PTS; p += stride) {
        int n = p >> 10, k = p & (N_PTS - 1);
        W1T[p] = (_Float16)W1[k * HID + n];
    }
    for (int p = idx; p < HID * HID; p += stride) {
        int n = p >> 7, k = p & (HID - 1);
        W2T[p] = (_Float16)W2[k * HID + n];
    }
}

// -----------------------------------------------------------------------------
// Kernel 1: circular autocorrelation via WMMA.
//   c[b,k] = (1/N) * sum_d sum_j s_d[j] * s_d[(j+k) mod N]
// Tile mapping: C[64m + n0 + n] = sum_j A[m,j]*B[j,n],
//   A[m,j] = s[j + 64m], B[j,n] = s[j - n0 - n]   (indices mod N)
// x[b] is bulk-copied to LDS with global_load_async_to_lds_b128 (ASYNCcnt),
// then converted to a replicated f16 signal (front pad 64, wrap tail) so the
// hot loop has no modulo math: A-fragment = 2x ds_load_b128, B-fragment =
// 9 aligned dwords + 8x v_alignbit (per-lane odd/even offset).
// -----------------------------------------------------------------------------
__global__ __launch_bounds__(256)
void corr_kernel(const float* __restrict__ x, _Float16* __restrict__ cH) {
    __shared__ __align__(16) float    xstage[N_PTS * 3];   // 12 KB raw x[b]
    __shared__ __align__(16) _Float16 sRep[3][SREP];
    __shared__ float cacc[2][N_PTS];

    const int b   = blockIdx.x;
    const int tid = threadIdx.x;

    // ---- async bulk copy: global x[b] -> LDS staging (3 x b128 per lane) ----
    {
        const char* src = (const char*)(x + (size_t)b * N_PTS * 3);
        const unsigned dst0 = lds_off32(&xstage[0]);
#pragma unroll
        for (int it = 0; it < 3; ++it) {
            const int byteoff = (tid + it * 256) * 16;
            unsigned ldsa = dst0 + (unsigned)byteoff;
            unsigned long long ga = (unsigned long long)(src + byteoff);
            asm volatile("global_load_async_to_lds_b128 %0, %1, off"
                         :: "v"(ldsa), "v"(ga) : "memory");
        }
        asm volatile("s_wait_asynccnt 0" ::: "memory");
    }
    __syncthreads();

    // ---- f32 -> f16 conversion with circular replication (reads hit LDS) ----
    for (int p = tid; p < SREP; p += 256) {
        int j3 = ((p + (N_PTS - 64)) & (N_PTS - 1)) * 3;
        sRep[0][p] = (_Float16)xstage[j3 + 0];
        sRep[1][p] = (_Float16)xstage[j3 + 1];
        sRep[2][p] = (_Float16)xstage[j3 + 2];
    }
    __syncthreads();

    const int w    = tid >> 5;
    const int lane = tid & 31;
    const int hi   = lane >> 4;
    const int l4   = lane & 15;
    const int t    = w & 3;          // n0 tile
    const int half = w >> 2;         // K half
    const int n0   = 16 * t;
    const int jbase = half * 512;

    // A: 16B-aligned (64|jbase|64*l4|16*hi all multiples of 8 halves)
    const int offA  = 64 + jbase + 64 * l4 + 16 * hi;
    // B: 16 consecutive halves at arbitrary even-byte offset
    const int offB  = 64 + jbase + 16 * hi - n0 - l4;
    const unsigned sh = (unsigned)(offB & 1) * 16u;
    const int offBe = offB & ~1;

    v8f acc = {};
    for (int d = 0; d < 3; ++d) {
        const _Float16* pA = &sRep[d][offA];
        const unsigned* pB = (const unsigned*)&sRep[d][offBe];
        for (int kk = 0; kk < 16; ++kk) {
            Frag a, bb;
            a.q[0] = *(const uint4*)(pA + 32 * kk);
            a.q[1] = *(const uint4*)(pA + 32 * kk + 8);
            unsigned wd[9];
#pragma unroll
            for (int i = 0; i < 9; ++i) wd[i] = pB[16 * kk + i];
#pragma unroll
            for (int c = 0; c < 8; ++c)
                bb.u[c] = __builtin_amdgcn_alignbit(wd[c + 1], wd[c], sh);
            acc = __builtin_amdgcn_wmma_f32_16x16x32_f16(
                false, a.h, false, bb.h, (short)0, acc, false, false);
        }
    }

    // D layout: VGPR r -> M = r + 8*hi, N = l4; k = 64*M + n0 + l4.
#pragma unroll
    for (int r = 0; r < 8; ++r) {
        int k = 64 * (r + 8 * hi) + n0 + l4;
        cacc[half][k] = acc[r];
    }
    __syncthreads();

    _Float16* co = cH + (size_t)b * N_PTS;
    for (int k = tid; k < N_PTS; k += 256)
        co[k] = (_Float16)((cacc[0][k] + cacc[1][k]) * (1.0f / (float)N_PTS));
}

// -----------------------------------------------------------------------------
// Kernel 2: fused MLP  y = celu(celu(c@W1+b1)@W2+b2)@W3 + b3
// One block, 8 waves (M = batch, one 16-row tile per wave). All WMMA operands
// are pairs of b128 loads: A from f16 cH / LDS h1 rows, B from transposed f16
// weight rows. h1, h2 live in LDS.
// -----------------------------------------------------------------------------
__global__ __launch_bounds__(256)
void mlp_kernel(const _Float16* __restrict__ cH,
                const _Float16* __restrict__ W1T, const float* __restrict__ b1,
                const _Float16* __restrict__ W2T, const float* __restrict__ b2,
                const float* __restrict__ W3, const float* __restrict__ b3,
                float* __restrict__ y) {
    __shared__ __align__(16) _Float16 h1[BATCH * HID];
    __shared__ __align__(16) _Float16 h2[BATCH * HID];

    const int tid  = threadIdx.x;
    const int w    = tid >> 5;
    const int lane = tid & 31;
    const int hi   = lane >> 4;
    const int l4   = lane & 15;
    const int mt   = w;

    // ---- Stage A: h1 = celu(c @ W1 + b1), K = 1024 ----
    const _Float16* aRow = cH + (size_t)(mt * 16 + l4) * N_PTS + 8 * hi;
    for (int nt = 0; nt < 8; ++nt) {
        const _Float16* bRow = W1T + (size_t)(nt * 16 + l4) * N_PTS + 8 * hi;
        v8f acc = {};
        for (int kk = 0; kk < 32; ++kk) {
            Frag a, bb;
            a.q[0]  = *(const uint4*)(aRow + 32 * kk);        // K groups with gap
            a.q[1]  = *(const uint4*)(aRow + 32 * kk + 16);
            bb.q[0] = *(const uint4*)(bRow + 32 * kk);        // K groups with gap
            bb.q[1] = *(const uint4*)(bRow + 32 * kk + 16);
            acc = __builtin_amdgcn_wmma_f32_16x16x32_f16(
                false, a.h, false, bb.h, (short)0, acc, false, false);
        }
        float bias = b1[nt * 16 + l4];
#pragma unroll
        for (int r = 0; r < 8; ++r) {
            int row = mt * 16 + r + 8 * hi;
            h1[row * HID + nt * 16 + l4] = (_Float16)celu1(acc[r] + bias);
        }
    }
    __syncthreads();

    // ---- Stage B: h2 = celu(h1 @ W2 + b2), K = 128 ----
    const _Float16* aRow2 = h1 + (mt * 16 + l4) * HID + 8 * hi;
    for (int nt = 0; nt < 8; ++nt) {
        const _Float16* bRow = W2T + (size_t)(nt * 16 + l4) * HID + 8 * hi;
        v8f acc = {};
#pragma unroll
        for (int kk = 0; kk < 4; ++kk) {
            Frag a, bb;
            a.q[0]  = *(const uint4*)(aRow2 + 32 * kk);
            a.q[1]  = *(const uint4*)(aRow2 + 32 * kk + 16);
            bb.q[0] = *(const uint4*)(bRow + 32 * kk);
            bb.q[1] = *(const uint4*)(bRow + 32 * kk + 16);
            acc = __builtin_amdgcn_wmma_f32_16x16x32_f16(
                false, a.h, false, bb.h, (short)0, acc, false, false);
        }
        float bias = b2[nt * 16 + l4];
#pragma unroll
        for (int r = 0; r < 8; ++r) {
            int row = mt * 16 + r + 8 * hi;
            h2[row * HID + nt * 16 + l4] = (_Float16)celu1(acc[r] + bias);
        }
    }
    __syncthreads();

    // ---- Stage C: y = h2 @ W3 + b3 (128x128 matvec, VALU) ----
    if (tid < BATCH) {
        float acc = b3[0];
#pragma unroll 8
        for (int k = 0; k < HID; ++k)
            acc += (float)h2[tid * HID + k] * W3[k];
        y[tid] = acc;
    }
}

// -----------------------------------------------------------------------------
extern "C" void kernel_launch(void* const* d_in, const int* in_sizes, int n_in,
                              void* d_out, int out_size, void* d_ws, size_t ws_size,
                              hipStream_t stream) {
    const float* x  = (const float*)d_in[0];
    const float* W1 = (const float*)d_in[1];
    const float* b1 = (const float*)d_in[2];
    const float* W2 = (const float*)d_in[3];
    const float* b2 = (const float*)d_in[4];
    const float* W3 = (const float*)d_in[5];
    const float* b3 = (const float*)d_in[6];
    float* out = (float*)d_out;

    // Workspace layout (f16): cH 256KB | W1T 256KB | W2T 32KB
    _Float16* cH  = (_Float16*)d_ws;
    _Float16* W1T = (_Float16*)((char*)d_ws + 262144);
    _Float16* W2T = (_Float16*)((char*)d_ws + 524288);

    prep_kernel<<<64, 256, 0, stream>>>(W1, W2, W1T, W2T);
    corr_kernel<<<BATCH, 256, 0, stream>>>(x, cH);
    mlp_kernel<<<1, 256, 0, stream>>>(cH, W1T, b1, W2T, b2, W3, b3, out);
}
